// EncoderDecoder3_35897336660442
// MI455X (gfx1250) — compile-verified, matrix operand order checked by reading, not run on hardware
//
#include <hip/hip_runtime.h>
#include <stdint.h>

// ---------------------------------------------------------------------------
// Types for WMMA operands
// ---------------------------------------------------------------------------
typedef __attribute__((ext_vector_type(16))) __bf16 v16bf;
typedef __attribute__((ext_vector_type(8)))  float  v8f;
typedef __attribute__((ext_vector_type(4)))  unsigned int v4u;
typedef __attribute__((ext_vector_type(4)))  float  v4f;

union ABReg { v4u u[2]; v16bf v; };

__device__ __forceinline__ unsigned short f2bf(float f) {
  unsigned int u = __float_as_uint(f);
  unsigned int r = u + 0x7FFFu + ((u >> 16) & 1u);   // round-to-nearest-even
  return (unsigned short)(r >> 16);
}

// ---------------------------------------------------------------------------
// Weight pre-pack: W[K,M] fp32 row-major -> bf16, per-lane WMMA B-operand order
// Wp[((kt*NT + nt)*32 + lane)*16 + j], klocal = (j<8 ? j : j+8) + (lane>=16)*8
// ---------------------------------------------------------------------------
__global__ void pack_w_kernel(const float* __restrict__ W,
                              unsigned short* __restrict__ Wp,
                              int K, int M) {
  int idx = blockIdx.x * blockDim.x + threadIdx.x;
  if (idx >= K * M) return;
  const int NT = M >> 4;
  int j    = idx & 15;
  int lane = (idx >> 4) & 31;
  int tile = idx >> 9;              // kt*NT + nt
  int nt   = tile % NT;
  int kt   = tile / NT;
  int klocal = (j < 8 ? j : j + 8) + ((lane & 16) ? 8 : 0);
  int k = (kt << 5) + klocal;
  int n = (nt << 4) + (lane & 15);
  Wp[idx] = f2bf(W[(size_t)k * M + n]);
}

// ---------------------------------------------------------------------------
// Edge-attr scatter to both endpoints + incident counts + in-degree (for GCN)
// ---------------------------------------------------------------------------
__global__ void edge_scatter_kernel(const int* __restrict__ ei,
                                    const float* __restrict__ ea,
                                    float* __restrict__ na,
                                    float* __restrict__ cntf,
                                    float* __restrict__ degf,
                                    int E) {
  long long t = (long long)blockIdx.x * blockDim.x + threadIdx.x;
  if (t >= (long long)E * 64) return;
  int e = (int)(t >> 6);
  int f = (int)(t & 63);
  float v = ea[(size_t)e * 64 + f];
  int r = ei[e];
  int c = ei[E + e];
  atomicAdd(na + (size_t)r * 64 + f, v);
  atomicAdd(na + (size_t)c * 64 + f, v);
  if (f == 0) {
    atomicAdd(cntf + r, 1.0f);
    atomicAdd(cntf + c, 1.0f);
    atomicAdd(degf + c, 1.0f);      // GCN norm: in-degree at target
  }
}

__global__ void dinv_kernel(const float* __restrict__ degf,
                            float* __restrict__ dinv, int N) {
  int v = blockIdx.x * blockDim.x + threadIdx.x;
  if (v >= N) return;
  dinv[v] = rsqrtf(degf[v] + 1.0f); // +1 self loop; always > 0
}

// h0 = concat(x, na/(cnt+1e-8)) converted to bf16, [N,128]
__global__ void build_h0_kernel(const float* __restrict__ x,
                                const float* __restrict__ na,
                                const float* __restrict__ cntf,
                                unsigned short* __restrict__ h0,
                                int N) {
  long long t = (long long)blockIdx.x * blockDim.x + threadIdx.x;
  if (t >= (long long)N * 64) return;
  int v = (int)(t >> 6);
  int f = (int)(t & 63);
  float xn = x[(size_t)v * 64 + f];
  float nn = na[(size_t)v * 64 + f] / (cntf[v] + 1e-8f);
  h0[(size_t)v * 128 + f]      = f2bf(xn);
  h0[(size_t)v * 128 + 64 + f] = f2bf(nn);
}

// ---------------------------------------------------------------------------
// GEMM: XW[N,M] = Xb[N,K](bf16) * Wp(K,M packed bf16), fp32 accumulate.
// One wave per 16-row block; nt outer, kt inner; v_wmma_f32_16x16x32_bf16.
// ---------------------------------------------------------------------------
__global__ void gcn_gemm_kernel(const unsigned short* __restrict__ Xb,
                                const unsigned short* __restrict__ Wp,
                                float* __restrict__ XW,
                                int N, int K, int M) {
  const int lane = threadIdx.x & 31;
  const int wave = (blockIdx.x * blockDim.x + threadIdx.x) >> 5;
  const int nblk = N >> 4;
  if (wave >= nblk) return;            // uniform per wave: EXEC stays all-ones

  const int KT = K >> 5;
  const int NT = M >> 4;
  const int r    = (wave << 4) + (lane & 15);
  const int off  = (lane >> 4) << 3;   // 0 or 8 (K sub-block per half-wave)
  const int m0   = (lane >> 4) << 3;   // C rows 0..7 or 8..15
  const int ncol = lane & 15;
  const unsigned short* xrow = Xb + (size_t)r * K;

  for (int nt = 0; nt < NT; ++nt) {
    v8f c = {0.f, 0.f, 0.f, 0.f, 0.f, 0.f, 0.f, 0.f};
    const unsigned short* bp = Wp + ((size_t)nt * 32 + lane) * 16;
    for (int kt = 0; kt < KT; ++kt) {
      ABReg a, b;
      const unsigned short* ap = xrow + (kt << 5) + off;
      a.u[0] = *(const v4u*)(ap);
      a.u[1] = *(const v4u*)(ap + 16);
      const unsigned short* bq = bp + (size_t)kt * NT * 512;
      b.u[0] = *(const v4u*)(bq);
      b.u[1] = *(const v4u*)(bq + 8);
      __builtin_prefetch(ap + 32, 0, 0);   // global_prefetch_b8 next K tile
      c = __builtin_amdgcn_wmma_f32_16x16x32_bf16(
              false, a.v, false, b.v, (short)0, c, false, false);
    }
    float* out = XW + (size_t)((wave << 4) + m0) * M + (nt << 4) + ncol;
#pragma unroll
    for (int i = 0; i < 8; ++i) out[(size_t)i * M] = c[i];
  }
}

// ---------------------------------------------------------------------------
// Edge aggregation: acc[col] += xw[row] * dinv[row]*dinv[col]   (fp32 atomics)
// ---------------------------------------------------------------------------
__global__ void agg_kernel(const float* __restrict__ xw,
                           const int* __restrict__ ei,
                           const float* __restrict__ dinv,
                           float* __restrict__ acc,
                           int M, int E) {
  long long t = (long long)blockIdx.x * blockDim.x + threadIdx.x;
  const int M4 = M >> 2;
  if (t >= (long long)E * M4) return;
  int e  = (int)(t / M4);
  int c4 = (int)(t % M4) << 2;
  int r  = ei[e];
  int cl = ei[E + e];
  float norm = dinv[r] * dinv[cl];
  v4f v = *(const v4f*)(xw + (size_t)r * M + c4);
  float* dst = acc + (size_t)cl * M + c4;
  atomicAdd(dst + 0, v[0] * norm);
  atomicAdd(dst + 1, v[1] * norm);
  atomicAdd(dst + 2, v[2] * norm);
  atomicAdd(dst + 3, v[3] * norm);
}

// Self-loop + bias + optional ReLU; emit bf16 (next layer) and/or fp32 (final)
__global__ void finalize_kernel(const float* __restrict__ acc,
                                const float* __restrict__ xw,
                                const float* __restrict__ dinv,
                                const float* __restrict__ bias,
                                unsigned short* __restrict__ outb,
                                float* __restrict__ outf,
                                int N, int M, int do_relu) {
  long long t = (long long)blockIdx.x * blockDim.x + threadIdx.x;
  if (t >= (long long)N * M) return;
  int v = (int)(t / M);
  int m = (int)(t % M);
  float di = dinv[v];
  float val = acc[t] + xw[t] * di * di + bias[m];
  if (do_relu) val = fmaxf(val, 0.0f);
  if (outb) outb[t] = f2bf(val);
  if (outf) outf[t] = val;
}

// ---------------------------------------------------------------------------
// Host launcher
// ---------------------------------------------------------------------------
extern "C" void kernel_launch(void* const* d_in, const int* in_sizes, int n_in,
                              void* d_out, int out_size, void* d_ws, size_t ws_size,
                              hipStream_t stream) {
  const float* x   = (const float*)d_in[0];
  const int*   ei  = (const int*)  d_in[1];
  const float* ea  = (const float*)d_in[2];
  const float* W1  = (const float*)d_in[3];
  const float* b1  = (const float*)d_in[4];
  const float* W2  = (const float*)d_in[5];
  const float* b2  = (const float*)d_in[6];
  const float* Wd1 = (const float*)d_in[7];
  const float* bd1 = (const float*)d_in[8];
  const float* Wd2 = (const float*)d_in[9];
  const float* bd2 = (const float*)d_in[10];
  float* out = (float*)d_out;

  const int N = in_sizes[0] / 64;      // 50000
  const int E = in_sizes[1] / 2;       // 800000

  // Workspace carve-out
  char* p = (char*)d_ws;
  auto alloc = [&](size_t bytes) -> char* {
    char* r = p; p += (bytes + 255) & ~(size_t)255; return r;
  };
  unsigned short* Wp1 = (unsigned short*)alloc((size_t)128 * 256 * 2);
  unsigned short* Wp2 = (unsigned short*)alloc((size_t)256 * 128 * 2);
  unsigned short* Wp3 = (unsigned short*)alloc((size_t)128 * 256 * 2);
  unsigned short* Wp4 = (unsigned short*)alloc((size_t)256 * 128 * 2);
  float* degf = (float*)alloc((size_t)N * 4);
  float* cntf = (float*)alloc((size_t)N * 4);
  float* dinv = (float*)alloc((size_t)N * 4);
  float* na   = (float*)alloc((size_t)N * 64 * 4);
  unsigned short* hA = (unsigned short*)alloc((size_t)N * 256 * 2);
  unsigned short* hB = (unsigned short*)alloc((size_t)N * 256 * 2);
  float* xw   = (float*)alloc((size_t)N * 256 * 4);
  float* acc  = (float*)alloc((size_t)N * 256 * 4);

  const int T = 256;
  auto blocks = [](long long total, int t) { return (unsigned)((total + t - 1) / t); };

  // --- weight packing ---
  pack_w_kernel<<<blocks(128 * 256, T), T, 0, stream>>>(W1,  Wp1, 128, 256);
  pack_w_kernel<<<blocks(256 * 128, T), T, 0, stream>>>(W2,  Wp2, 256, 128);
  pack_w_kernel<<<blocks(128 * 256, T), T, 0, stream>>>(Wd1, Wp3, 128, 256);
  pack_w_kernel<<<blocks(256 * 128, T), T, 0, stream>>>(Wd2, Wp4, 256, 128);

  // --- edge attr -> node attr, degrees ---
  hipMemsetAsync(na,   0, (size_t)N * 64 * 4, stream);
  hipMemsetAsync(cntf, 0, (size_t)N * 4, stream);
  hipMemsetAsync(degf, 0, (size_t)N * 4, stream);
  edge_scatter_kernel<<<blocks((long long)E * 64, T), T, 0, stream>>>(ei, ea, na, cntf, degf, E);
  dinv_kernel<<<blocks(N, T), T, 0, stream>>>(degf, dinv, N);
  build_h0_kernel<<<blocks((long long)N * 64, T), T, 0, stream>>>(x, na, cntf, hA, N);

  // --- 4 GCN layers ---
  struct Layer {
    const unsigned short* Xb; unsigned short* Ob; float* Of;
    const unsigned short* Wp; const float* bias; int K, M, relu;
  } layers[4] = {
    { hA, hB, nullptr, Wp1, b1,  128, 256, 1 },
    { hB, hA, nullptr, Wp2, b2,  256, 128, 1 },
    { hA, hB, nullptr, Wp3, bd1, 128, 256, 1 },
    { hB, nullptr, out, Wp4, bd2, 256, 128, 0 },
  };

  const unsigned gemm_blocks = blocks((long long)(N / 16) * 32, T);
  for (int L = 0; L < 4; ++L) {
    const Layer& ly = layers[L];
    gcn_gemm_kernel<<<gemm_blocks, T, 0, stream>>>(ly.Xb, ly.Wp, xw, N, ly.K, ly.M);
    hipMemsetAsync(acc, 0, (size_t)N * ly.M * 4, stream);
    agg_kernel<<<blocks((long long)E * (ly.M / 4), T), T, 0, stream>>>(xw, ei, dinv, acc, ly.M, E);
    finalize_kernel<<<blocks((long long)N * ly.M, T), T, 0, stream>>>(
        acc, xw, dinv, ly.bias, ly.Ob, ly.Of, N, ly.M, ly.relu);
  }
}
